// StreamTGN_55722905699033
// MI455X (gfx1250) — matrix operand order repaired and checked by reading, not compiled, not run on hardware
//
#include <hip/hip_runtime.h>

#define NN   500000
#define BQ   8192
#define KN   32
#define DM   128
#define TT   64
#define KSEL 16

typedef __bf16 bf16_t;
typedef __attribute__((ext_vector_type(16))) __bf16 v16bf;
typedef __attribute__((ext_vector_type(8)))  float  v8f;

union FragAB { v16bf v; unsigned int u[8]; };

// A fragment (16x32 bf16, M=row): lane L: m=L&15, half=L>>4.
// VGPR v in 0..3 -> K = half*8 + 2v ; v in 4..7 -> K = 16 + half*8 + 2(v-4)
static __device__ inline void load_a_frag(const bf16_t* base, int ldk, FragAB& f) {
  int lane = threadIdx.x & 31;
  int m = lane & 15;
  int hf = lane >> 4;
  const bf16_t* row = base + m * ldk;
#pragma unroll
  for (int v = 0; v < 4; ++v)
    f.u[v] = *(const unsigned int*)(row + hf * 8 + 2 * v);
#pragma unroll
  for (int v = 0; v < 4; ++v)
    f.u[4 + v] = *(const unsigned int*)(row + 16 + hf * 8 + 2 * v);
}

// B fragment (32x16 bf16, B[k][n] = W[n][k], W row-major [out][in]):
// lane L: n = n0 + (L&15), kbase = k0 + (L>=16 ? 8 : 0); VGPR v in 0..3 -> k = kbase + 2v ; v in 4..7 -> k = kbase + 16 + 2(v-4)
static __device__ inline void load_b_frag(const bf16_t* W, int ldk, int n0, int k0, FragAB& f) {
  int lane = threadIdx.x & 31;
  int n = n0 + (lane & 15);
  int kb = k0 + ((lane >> 4) << 3);
  const bf16_t* row = W + n * ldk + kb;
#pragma unroll
  for (int v = 0; v < 4; ++v)
    f.u[v] = *(const unsigned int*)(row + 2 * v);
#pragma unroll
  for (int v = 0; v < 4; ++v)
    f.u[4 + v] = *(const unsigned int*)(row + 16 + 2 * v);
}

static __device__ inline v8f wmma_bf16(const FragAB& a, const FragAB& b, v8f c) {
  return __builtin_amdgcn_wmma_f32_16x16x32_bf16(false, a.v, false, b.v, (short)0, c, false, false);
}

// ---------------- utility kernels ----------------
__global__ void k_cvt_bf16(const float* __restrict__ s, bf16_t* __restrict__ d, int n) {
  int i = blockIdx.x * blockDim.x + threadIdx.x;
  if (i < n) d[i] = (bf16_t)s[i];
}

__global__ void k_init_idx(int* __restrict__ m, int n) {
  int i = blockIdx.x * blockDim.x + threadIdx.x;
  if (i < n) m[i] = -1;
}

__global__ void k_scatter_idx(const int* __restrict__ src, const int* __restrict__ dst,
                              int* __restrict__ m) {
  int i = blockIdx.x * blockDim.x + threadIdx.x;
  if (i < 2 * BQ) {
    int node = (i < BQ) ? src[i] : dst[i - BQ];
    atomicMax(&m[node], i);   // last occurrence wins (dst half > src half)
  }
}

// ---------------- stage 1: message MLP + GRU ----------------
// block = 64 threads (2 waves), 32 rows per block; wave handles 16 rows
__global__ __launch_bounds__(64) void k_stage1(
    const int* __restrict__ src, const int* __restrict__ dst, const float* __restrict__ ts,
    const float* __restrict__ memory, const float* __restrict__ last_update,
    const float* __restrict__ w_t_msg, const float* __restrict__ b_t_msg,
    const float* __restrict__ b_msg1, const float* __restrict__ b_msg2,
    const float* __restrict__ b_ih, const float* __restrict__ b_hh,
    const bf16_t* __restrict__ Wmsg1b, const bf16_t* __restrict__ Wmsg2b,
    const bf16_t* __restrict__ Wihb, const bf16_t* __restrict__ Whhb,
    float* __restrict__ new_mem)
{
  __shared__ bf16_t Xs[32 * 320];   // [self 0:128 | other 128:256 | te 256:320]
  __shared__ bf16_t Hs[32 * 128];
  __shared__ bf16_t Ms[32 * 128];
  __shared__ int   s_node[32];
  __shared__ int   s_oth[32];
  __shared__ float s_dt[32];

  int tid = threadIdx.x;
  int r0 = blockIdx.x * 32;

  if (tid < 32) {
    int i = r0 + tid;
    int node, other; float tb;
    if (i < BQ) { node = src[i]; other = dst[i]; tb = ts[i]; }
    else        { node = dst[i - BQ]; other = src[i - BQ]; tb = ts[i - BQ]; }
    s_node[tid] = node; s_oth[tid] = other;
    s_dt[tid] = tb - last_update[node];
  }
  __syncthreads();

  for (int idx = tid; idx < 32 * 320; idx += 64) {
    int lr = idx / 320, c = idx % 320;
    float v;
    if (c < 128)       v = memory[s_node[lr] * DM + c];
    else if (c < 256)  v = memory[s_oth[lr] * DM + (c - 128)];
    else {
      int t = c - 256;
      float f = s_dt[lr] * w_t_msg[t] + b_t_msg[t];
      v = (t & 1) ? __cosf(f) : __sinf(f);
    }
    Xs[idx] = (bf16_t)v;
  }
  __syncthreads();

  int wave = tid >> 5, lane = tid & 31, ml = lane & 15, hf = lane >> 4;
  int wr = wave * 16;

  // hidden = relu(X @ Wmsg1^T + b1)  K=320
#pragma unroll 1
  for (int t = 0; t < 8; ++t) {
    v8f acc; float bias = b_msg1[t * 16 + ml];
#pragma unroll
    for (int vg = 0; vg < 8; ++vg) acc[vg] = bias;
    for (int kc = 0; kc < 10; ++kc) {
      FragAB a, b;
      load_a_frag(&Xs[wr * 320 + kc * 32], 320, a);
      load_b_frag(Wmsg1b, 320, t * 16, kc * 32, b);
      acc = wmma_bf16(a, b, acc);
    }
#pragma unroll
    for (int vg = 0; vg < 8; ++vg) {
      int m = vg + 8 * hf;
      Hs[(wr + m) * 128 + t * 16 + ml] = (bf16_t)fmaxf(acc[vg], 0.0f);
    }
  }

  // msg = hidden @ Wmsg2^T + b2   (own rows -> same-wave LDS ordering, no barrier)
#pragma unroll 1
  for (int t = 0; t < 8; ++t) {
    v8f acc; float bias = b_msg2[t * 16 + ml];
#pragma unroll
    for (int vg = 0; vg < 8; ++vg) acc[vg] = bias;
    for (int kc = 0; kc < 4; ++kc) {
      FragAB a, b;
      load_a_frag(&Hs[wr * 128 + kc * 32], 128, a);
      load_b_frag(Wmsg2b, 128, t * 16, kc * 32, b);
      acc = wmma_bf16(a, b, acc);
    }
#pragma unroll
    for (int vg = 0; vg < 8; ++vg) {
      int m = vg + 8 * hf;
      Ms[(wr + m) * 128 + t * 16 + ml] = (bf16_t)acc[vg];
    }
  }

  // GRU: gi = msg @ Wih^T + b_ih ; gh = self @ Whh^T + b_hh  (r/z/g sections)
#pragma unroll 1
  for (int t = 0; t < 8; ++t) {
    int n = t * 16 + ml;
    v8f ir, iz, ig, hr, hz, hg;
    {
      float b0 = b_ih[n], b1 = b_ih[128 + n], b2 = b_ih[256 + n];
      float c0 = b_hh[n], c1 = b_hh[128 + n], c2 = b_hh[256 + n];
#pragma unroll
      for (int vg = 0; vg < 8; ++vg) {
        ir[vg] = b0; iz[vg] = b1; ig[vg] = b2;
        hr[vg] = c0; hz[vg] = c1; hg[vg] = c2;
      }
    }
    for (int kc = 0; kc < 4; ++kc) {
      FragAB am, as, b;
      load_a_frag(&Ms[wr * 128 + kc * 32], 128, am);
      load_a_frag(&Xs[wr * 320 + kc * 32], 320, as);   // self mem cols 0..127
      load_b_frag(Wihb, 128, t * 16, kc * 32, b);        ir = wmma_bf16(am, b, ir);
      load_b_frag(Wihb, 128, 128 + t * 16, kc * 32, b);  iz = wmma_bf16(am, b, iz);
      load_b_frag(Wihb, 128, 256 + t * 16, kc * 32, b);  ig = wmma_bf16(am, b, ig);
      load_b_frag(Whhb, 128, t * 16, kc * 32, b);        hr = wmma_bf16(as, b, hr);
      load_b_frag(Whhb, 128, 128 + t * 16, kc * 32, b);  hz = wmma_bf16(as, b, hz);
      load_b_frag(Whhb, 128, 256 + t * 16, kc * 32, b);  hg = wmma_bf16(as, b, hg);
    }
#pragma unroll
    for (int vg = 0; vg < 8; ++vg) {
      int m = vg + 8 * hf;
      int col = t * 16 + ml;
      float r = 1.0f / (1.0f + __expf(-(ir[vg] + hr[vg])));
      float z = 1.0f / (1.0f + __expf(-(iz[vg] + hz[vg])));
      float g = tanhf(ig[vg] + r * hg[vg]);
      float sm = (float)Xs[(wr + m) * 320 + col];
      new_mem[(r0 + wr + m) * DM + col] = (1.0f - z) * g + z * sm;
    }
  }
}

// ---------------- top-k temporal neighbor sampling ----------------
__global__ void k_topk(const int* __restrict__ src, const float* __restrict__ ts,
                       const int* __restrict__ nbr_ids, const float* __restrict__ nbr_ts,
                       int* __restrict__ sel, float* __restrict__ td, int* __restrict__ mkb)
{
  int b = blockIdx.x * blockDim.x + threadIdx.x;
  if (b >= BQ) return;
  int s = src[b]; float t = ts[b];
  float key[KN]; int id[KN];
#pragma unroll
  for (int j = 0; j < KN; ++j) {
    int v = nbr_ids[s * KN + j];
    float nt = nbr_ts[s * KN + j];
    bool valid = (v >= 0) && (nt < t + 1e-6f);
    id[j] = v;
    key[j] = valid ? nt : -1e30f;
  }
  unsigned taken = 0;
  for (int slot = 0; slot < KSEL; ++slot) {
    float best = -3.4e38f; int bj = 0;
    for (int j = 0; j < KN; ++j) {
      if (!((taken >> j) & 1u) && key[j] > best) { best = key[j]; bj = j; }
    }
    taken |= (1u << bj);
    bool mk = key[bj] > -1.0e29f;
    int sid = mk ? id[bj] : 0;
    sid = sid < 0 ? 0 : (sid > NN - 1 ? NN - 1 : sid);
    sel[b * KSEL + slot] = sid;
    td[b * KSEL + slot]  = mk ? (t - key[bj]) : 0.0f;
    mkb[b * KSEL + slot] = mk ? 1 : 0;
  }
}

// ---------------- node features + Q ----------------
__global__ __launch_bounds__(128) void k_qnode(
    const int* __restrict__ src, const float* __restrict__ memory,
    const int* __restrict__ idx_map, const float* __restrict__ new_mem,
    const float* __restrict__ bq, const bf16_t* __restrict__ Wqb,
    float* __restrict__ node_feat_g, float* __restrict__ Qg)
{
  __shared__ bf16_t NF[64 * 128];
  int tid = threadIdx.x;
  int r0 = blockIdx.x * 64;
  for (int idx = tid; idx < 64 * 128; idx += 128) {
    int lr = idx >> 7, c = idx & 127;
    int b = r0 + lr;
    int node = src[b];
    int j = idx_map[node];
    float v = (j >= 0) ? new_mem[j * DM + c] : memory[node * DM + c];
    node_feat_g[b * DM + c] = v;
    NF[idx] = (bf16_t)v;
  }
  __syncthreads();
  int wave = tid >> 5, lane = tid & 31, ml = lane & 15, hf = lane >> 4;
  int wr = wave * 16;
#pragma unroll 1
  for (int t = 0; t < 8; ++t) {
    v8f acc; float bias = bq[t * 16 + ml];
#pragma unroll
    for (int vg = 0; vg < 8; ++vg) acc[vg] = bias;
    for (int kc = 0; kc < 4; ++kc) {
      FragAB a, b;
      load_a_frag(&NF[wr * 128 + kc * 32], 128, a);
      load_b_frag(Wqb, 128, t * 16, kc * 32, b);
      acc = wmma_bf16(a, b, acc);
    }
#pragma unroll
    for (int vg = 0; vg < 8; ++vg) {
      int m = vg + 8 * hf;
      Qg[(r0 + wr + m) * DM + t * 16 + ml] = acc[vg];
    }
  }
}

// ---------------- attention (wave per query, 4 queries per block) ----------------
__global__ __launch_bounds__(128) void k_attn(
    const int* __restrict__ sel, const float* __restrict__ td, const int* __restrict__ mkb,
    const int* __restrict__ idx_map, const float* __restrict__ new_mem,
    const float* __restrict__ memory,
    const float* __restrict__ w_t_att, const float* __restrict__ b_t_att,
    const float* __restrict__ Qg, const float* __restrict__ bk, const float* __restrict__ bv,
    const bf16_t* __restrict__ Wkb, const bf16_t* __restrict__ Wvb,
    const float* __restrict__ Wo, const float* __restrict__ bo,
    float* __restrict__ attn_out_g)
{
  __shared__ bf16_t KV[4 * 16 * 192];
  __shared__ float  Qs[4 * 128];
  __shared__ float  stg[4][16][16];
  __shared__ float  attw[4][2][16];
  __shared__ float  outb[4][128];

  int tid = threadIdx.x;
  int q0 = blockIdx.x * 4;

  for (int idx = tid; idx < 4 * 16 * 192; idx += 128) {
    int ql = idx / (16 * 192);
    int rem = idx % (16 * 192);
    int j = rem / 192, c = rem % 192;
    int b = q0 + ql;
    float v;
    if (c < 128) {
      int nid = sel[b * KSEL + j];
      int ii = idx_map[nid];
      v = (ii >= 0) ? new_mem[ii * DM + c] : memory[nid * DM + c];
    } else {
      int tt = c - 128;
      float f = td[b * KSEL + j] * w_t_att[tt] + b_t_att[tt];
      v = (tt & 1) ? __cosf(f) : __sinf(f);
    }
    KV[idx] = (bf16_t)v;
  }
  for (int idx = tid; idx < 4 * 128; idx += 128) {
    int ql = idx >> 7, c = idx & 127;
    Qs[idx] = Qg[(q0 + ql) * DM + c];
  }
  __syncthreads();

  int wave = tid >> 5, lane = tid & 31, ml = lane & 15, hf = lane >> 4;
  int b = q0 + wave;
  int j = ml, h = hf;   // 32 lanes = 16 neighbors x 2 heads

  // K phase: Kp tiles -> per-(j,h) scores
  float sc = 0.0f;
#pragma unroll 1
  for (int t = 0; t < 8; ++t) {
    v8f acc; float bias = bk[t * 16 + ml];
#pragma unroll
    for (int vg = 0; vg < 8; ++vg) acc[vg] = bias;
    for (int kc = 0; kc < 6; ++kc) {
      FragAB a, bb;
      load_a_frag(&KV[wave * 16 * 192 + kc * 32], 192, a);
      load_b_frag(Wkb, 192, t * 16, kc * 32, bb);
      acc = wmma_bf16(a, bb, acc);
    }
#pragma unroll
    for (int vg = 0; vg < 8; ++vg) stg[wave][vg + 8 * hf][ml] = acc[vg];
    if ((t >> 2) == h) {      // tile t belongs to head t/4; same-wave LDS is in-order
      for (int c = 0; c < 16; ++c) sc += stg[wave][j][c] * Qs[wave * 128 + t * 16 + c];
    }
  }
  sc *= 0.125f;   // 1/sqrt(64)
  int mk = mkb[b * KSEL + j];
  if (!mk) sc = -3.0e38f;
  float mx = sc;
#pragma unroll
  for (int o = 1; o < 16; o <<= 1) mx = fmaxf(mx, __shfl_xor(mx, o, 32));
  float e = mk ? __expf(sc - mx) : 0.0f;
  float ssum = e;
#pragma unroll
  for (int o = 1; o < 16; o <<= 1) ssum += __shfl_xor(ssum, o, 32);
  attw[wave][h][j] = (ssum > 0.0f) ? (e / ssum) : 0.0f;

  // V phase: Vp tiles -> weighted sum per output column
#pragma unroll 1
  for (int t = 0; t < 8; ++t) {
    v8f acc; float bias = bv[t * 16 + ml];
#pragma unroll
    for (int vg = 0; vg < 8; ++vg) acc[vg] = bias;
    for (int kc = 0; kc < 6; ++kc) {
      FragAB a, bb;
      load_a_frag(&KV[wave * 16 * 192 + kc * 32], 192, a);
      load_b_frag(Wvb, 192, t * 16, kc * 32, bb);
      acc = wmma_bf16(a, bb, acc);
    }
#pragma unroll
    for (int vg = 0; vg < 8; ++vg) stg[wave][vg + 8 * hf][ml] = acc[vg];
    if (lane < 16) {
      int c = lane, hh = t >> 2;
      float o = 0.0f;
      for (int jj = 0; jj < 16; ++jj) o += attw[wave][hh][jj] * stg[wave][jj][c];
      outb[wave][t * 16 + c] = o;
    }
  }
  __syncthreads();

  // attn_out = out @ Wo^T + bo  (tiny M: plain VALU with f32 weights)
  for (int idx = tid; idx < 4 * 128; idx += 128) {
    int ql = idx >> 7, n = idx & 127;
    float s = bo[n];
    for (int k2 = 0; k2 < 128; ++k2) s += outb[ql][k2] * Wo[n * 128 + k2];
    attn_out_g[(q0 + ql) * DM + n] = s;
  }
}

// ---------------- merge MLP + layernorm ----------------
__global__ __launch_bounds__(128) void k_merge(
    const float* __restrict__ attn_out_g, const float* __restrict__ node_feat_g,
    const float* __restrict__ bm1, const float* __restrict__ bm2,
    const bf16_t* __restrict__ Wm1b, const bf16_t* __restrict__ Wm2b,
    const float* __restrict__ ln_g, const float* __restrict__ ln_b,
    float* __restrict__ out)
{
  __shared__ bf16_t CB[64 * 256];
  __shared__ bf16_t H1[64 * 128];
  int tid = threadIdx.x;
  int r0 = blockIdx.x * 64;
  for (int idx = tid; idx < 64 * 256; idx += 128) {
    int lr = idx >> 8, c = idx & 255;
    int b = r0 + lr;
    float v = (c < 128) ? attn_out_g[b * DM + c] : node_feat_g[b * DM + (c - 128)];
    CB[idx] = (bf16_t)v;
  }
  __syncthreads();

  int wave = tid >> 5, lane = tid & 31, ml = lane & 15, hf = lane >> 4;
  int wr = wave * 16;

  // h1 = relu(comb @ Wm1^T + bm1)  K=256
#pragma unroll 1
  for (int t = 0; t < 8; ++t) {
    v8f acc; float bias = bm1[t * 16 + ml];
#pragma unroll
    for (int vg = 0; vg < 8; ++vg) acc[vg] = bias;
    for (int kc = 0; kc < 8; ++kc) {
      FragAB a, b;
      load_a_frag(&CB[wr * 256 + kc * 32], 256, a);
      load_b_frag(Wm1b, 256, t * 16, kc * 32, b);
      acc = wmma_bf16(a, b, acc);
    }
#pragma unroll
    for (int vg = 0; vg < 8; ++vg) {
      int m = vg + 8 * hf;
      H1[(wr + m) * 128 + t * 16 + ml] = (bf16_t)fmaxf(acc[vg], 0.0f);
    }
  }

  // h = h1 @ Wm2^T + bm2  -> keep all 8 tiles in registers for layernorm
  v8f hacc[8];
#pragma unroll 1
  for (int t = 0; t < 8; ++t) {
    float bias = bm2[t * 16 + ml];
#pragma unroll
    for (int vg = 0; vg < 8; ++vg) hacc[t][vg] = bias;
    for (int kc = 0; kc < 4; ++kc) {
      FragAB a, b;
      load_a_frag(&H1[wr * 128 + kc * 32], 128, a);
      load_b_frag(Wm2b, 128, t * 16, kc * 32, b);
      hacc[t] = wmma_bf16(a, b, hacc[t]);
    }
  }

  // layernorm per row: row m = vg + 8*half is held by 16 lanes of one half-wave
  float mu[8], inv[8];
#pragma unroll
  for (int vg = 0; vg < 8; ++vg) {
    float s = 0.0f, sq = 0.0f;
#pragma unroll
    for (int t = 0; t < 8; ++t) { float x = hacc[t][vg]; s += x; sq += x * x; }
#pragma unroll
    for (int o = 1; o < 16; o <<= 1) { s += __shfl_xor(s, o, 32); sq += __shfl_xor(sq, o, 32); }
    float m_ = s * (1.0f / 128.0f);
    float v_ = sq * (1.0f / 128.0f) - m_ * m_;
    mu[vg] = m_;
    inv[vg] = rsqrtf(v_ + 1e-5f);
  }
#pragma unroll
  for (int t = 0; t < 8; ++t) {
    int col = t * 16 + ml;
    float g = ln_g[col], bb = ln_b[col];
#pragma unroll
    for (int vg = 0; vg < 8; ++vg) {
      int m = vg + 8 * hf;
      out[(r0 + wr + m) * DM + col] = (hacc[t][vg] - mu[vg]) * inv[vg] * g + bb;
    }
  }
}

// ---------------- host launch ----------------
extern "C" void kernel_launch(void* const* d_in, const int* in_sizes, int n_in,
                              void* d_out, int out_size, void* d_ws, size_t ws_size,
                              hipStream_t stream) {
  (void)in_sizes; (void)n_in; (void)out_size; (void)ws_size;
  const int*   src  = (const int*)  d_in[0];
  const int*   dst  = (const int*)  d_in[1];
  const float* ts   = (const float*)d_in[2];
  const int*   nbr_ids = (const int*)d_in[3];
  const float* nbr_ts  = (const float*)d_in[4];
  const float* memory  = (const float*)d_in[5];
  const float* last_update = (const float*)d_in[6];
  const float* w_t_msg = (const float*)d_in[7];
  const float* b_t_msg = (const float*)d_in[8];
  const float* W_msg1  = (const float*)d_in[9];
  const float* b_msg1  = (const float*)d_in[10];
  const float* W_msg2  = (const float*)d_in[11];
  const float* b_msg2  = (const float*)d_in[12];
  const float* W_ih = (const float*)d_in[13];
  const float* b_ih = (const float*)d_in[14];
  const float* W_hh = (const float*)d_in[15];
  const float* b_hh = (const float*)d_in[16];
  const float* w_t_att = (const float*)d_in[17];
  const float* b_t_att = (const float*)d_in[18];
  const float* Wq = (const float*)d_in[19];
  const float* bq = (const float*)d_in[20];
  const float* Wk = (const float*)d_in[21];
  const float* bk = (const float*)d_in[22];
  const float* Wv = (const float*)d_in[23];
  const float* bv = (const float*)d_in[24];
  const float* Wo = (const float*)d_in[25];
  const float* bo = (const float*)d_in[26];
  const float* Wm1 = (const float*)d_in[27];
  const float* bm1 = (const float*)d_in[28];
  const float* Wm2 = (const float*)d_in[29];
  const float* bm2 = (const float*)d_in[30];
  const float* ln_g = (const float*)d_in[31];
  const float* ln_b = (const float*)d_in[32];
  float* out = (float*)d_out;

  char* ws = (char*)d_ws;
  size_t p = 0;
  auto carve = [&](size_t bytes) -> void* {
    void* r = ws + p;
    p += (bytes + 255) & ~(size_t)255;
    return r;
  };
  int*    idx_map = (int*)   carve((size_t)NN * 4);
  float*  new_mem = (float*) carve((size_t)2 * BQ * DM * 4);
  int*    selb    = (int*)   carve((size_t)BQ * KSEL * 4);
  float*  tdb     = (float*) carve((size_t)BQ * KSEL * 4);
  int*    mkbuf   = (int*)   carve((size_t)BQ * KSEL * 4);
  float*  Qb      = (float*) carve((size_t)BQ * DM * 4);
  float*  nfb     = (float*) carve((size_t)BQ * DM * 4);
  float*  aob     = (float*) carve((size_t)BQ * DM * 4);
  bf16_t* wbf     = (bf16_t*)carve((size_t)270336 * 2);

  bf16_t* Wmsg1b = wbf;                 // 128x320
  bf16_t* Wmsg2b = Wmsg1b + 40960;      // 128x128
  bf16_t* Wihb   = Wmsg2b + 16384;      // 384x128
  bf16_t* Whhb   = Wihb   + 49152;      // 384x128
  bf16_t* Wqb    = Whhb   + 49152;      // 128x128
  bf16_t* Wkb    = Wqb    + 16384;      // 128x192
  bf16_t* Wvb    = Wkb    + 24576;      // 128x192
  bf16_t* Wm1b   = Wvb    + 24576;      // 128x256
  bf16_t* Wm2b   = Wm1b   + 32768;      // 128x128

  k_init_idx<<<(NN + 255) / 256, 256, 0, stream>>>(idx_map, NN);

  struct { const float* s; bf16_t* d; int n; } cv[9] = {
    {W_msg1, Wmsg1b, 40960}, {W_msg2, Wmsg2b, 16384},
    {W_ih, Wihb, 49152}, {W_hh, Whhb, 49152},
    {Wq, Wqb, 16384}, {Wk, Wkb, 24576}, {Wv, Wvb, 24576},
    {Wm1, Wm1b, 32768}, {Wm2, Wm2b, 16384},
  };
  for (int i = 0; i < 9; ++i)
    k_cvt_bf16<<<(cv[i].n + 255) / 256, 256, 0, stream>>>(cv[i].s, cv[i].d, cv[i].n);

  k_scatter_idx<<<(2 * BQ + 255) / 256, 256, 0, stream>>>(src, dst, idx_map);

  k_stage1<<<(2 * BQ) / 32, 64, 0, stream>>>(
      src, dst, ts, memory, last_update, w_t_msg, b_t_msg,
      b_msg1, b_msg2, b_ih, b_hh, Wmsg1b, Wmsg2b, Wihb, Whhb, new_mem);

  k_topk<<<BQ / 256, 256, 0, stream>>>(src, ts, nbr_ids, nbr_ts, selb, tdb, mkbuf);

  k_qnode<<<BQ / 64, 128, 0, stream>>>(src, memory, idx_map, new_mem, bq, Wqb, nfb, Qb);

  k_attn<<<BQ / 4, 128, 0, stream>>>(
      selb, tdb, mkbuf, idx_map, new_mem, memory, w_t_att, b_t_att,
      Qb, bk, bv, Wkb, Wvb, Wo, bo, aob);

  k_merge<<<BQ / 64, 128, 0, stream>>>(aob, nfb, bm1, bm2, Wm1b, Wm2b, ln_g, ln_b, out);
}
